// HierarchicalWTABlock_27144193310734
// MI455X (gfx1250) — matrix operand order repaired and checked by reading, not compiled, not run on hardware
//
#include <hip/hip_runtime.h>
#include <hip/hip_bf16.h>
#include <math.h>

typedef _Float16 half_t;
typedef __attribute__((ext_vector_type(16))) _Float16 v16h;
typedef __attribute__((ext_vector_type(8)))  float    v8f;

#define BB 8
#define LL 4096
#define DD 256
#define GG 64
#define KK 16
#define NN 1024
#define D2 512

#if defined(__gfx1250__)
#define HAS_ASYNC_LDS 1
#else
#define HAS_ASYNC_LDS 0
#endif

#if HAS_ASYNC_LDS
// ISA §15.18 GLOBAL_LOAD_ASYNC_TO_LDS_B128: VDST = LDS byte address (VGPR),
// VADDR = 64-bit global address. Generic LDS pointer's low 32 bits are the LDS
// byte offset (flat aperture rule, ISA 10.2).
// The clang builtin wants a __device__-LangAS pointer we cannot spell in HIP,
// so use inline asm (tracked by ASYNCcnt).
__device__ __forceinline__ void async_copy_b128(const void* g, void* l) {
    unsigned int       lds_off = (unsigned int)(uintptr_t)l;
    unsigned long long gaddr   = (unsigned long long)(uintptr_t)g;
    asm volatile("global_load_async_to_lds_b128 %0, %1, off"
                 :: "v"(lds_off), "v"(gaddr) : "memory");
}
__device__ __forceinline__ void async_wait0() {
#if __has_builtin(__builtin_amdgcn_s_wait_asynccnt)
    __builtin_amdgcn_s_wait_asynccnt(0);
#else
    asm volatile("s_wait_asynccnt 0x0" ::: "memory");
#endif
}
#endif

__device__ __forceinline__ float gelu_exact(float x) {
    return 0.5f * x * (1.0f + erff(x * 0.70710678118654752440f));
}

// A-fragment (16x32 f16), contiguous K order matching load_b_frag:
// lanes 0-15: row M = lane, halves 0..15 -> K=k0..k0+15
// lanes 16-31: row M = lane-16, halves 0..15 -> K=k0+16..k0+31
__device__ __forceinline__ v16h load_a_frag(const half_t* base, int stride, int k0, int lane) {
    int row = lane & 15, hi = lane >> 4;
    const half_t* p0 = base + row * stride + k0 + hi * 16;
    v16h r;
#pragma unroll
    for (int e = 0; e < 16; ++e) { r[e] = p0[e]; }
    return r;
}

// B-fragment (32x16 f16): lane = column N, halves = consecutive K
// (lanes 16-31 cover K = k0+16..k0+31). Wt is f16 transposed [out][in],
// so this is ONE contiguous 32-byte load per lane.
__device__ __forceinline__ v16h load_b_frag(const half_t* Wt, int din, int n0, int k0, int lane) {
    int col = lane & 15, hi = lane >> 4;
    const half_t* p = Wt + (size_t)(n0 + col) * din + k0 + hi * 16;
    return *(const v16h*)p;
}

// fp32 (din,dout) row-major  ->  f16 transposed [dout][din]
__global__ void convw_kernel(const float* __restrict__ src, half_t* __restrict__ dst,
                             int din, int dout) {
    int i = blockIdx.x * 256 + threadIdx.x;
    if (i < din * dout) {
        int r = i / dout, c = i % dout;
        dst[(size_t)c * din + r] = (half_t)src[i];
    }
}

__global__ __launch_bounds__(256)
void wta_token_kernel(const float* __restrict__ X,
                      const float* __restrict__ Wgroup, const float* __restrict__ Wslot,
                      const float* __restrict__ bm1, const float* __restrict__ bm2,
                      const float* __restrict__ bg1, const float* __restrict__ bg2,
                      const float* __restrict__ Wg2,
                      const half_t* __restrict__ Wm1t, const half_t* __restrict__ Wm2t,
                      const half_t* __restrict__ Wg1t,
                      float* __restrict__ incoming) {
    __shared__ float  Xf[16][DD];        // fp32 X tile (routing + reused as Hg)
    __shared__ half_t Xh[16][DD + 16];   // f16 X tile (WMMA A)
    __shared__ half_t H1h[16][D2 + 16];  // gelu(X@Wm1+b) f16
    __shared__ float  Buf[16][DD];       // routing logits, then msg (fp32)
    __shared__ float  red[16][17];
    __shared__ float  gatev[16];
    __shared__ int    slotv[16];

    const int tid  = threadIdx.x;
    const int wave = tid >> 5, lane = tid & 31;
    const int arow = lane & 15, hi = lane >> 4;
    const size_t t0 = (size_t)blockIdx.x * 16;   // 16 tokens per block; L%16==0 so no batch straddle
    const int b = (int)(t0 / LL);

    // ---- stage X tile into LDS (async DMA when available), then f16 copy ----
#if HAS_ASYNC_LDS
    {
        const float* xg = X + t0 * DD;
#pragma unroll
        for (int j = 0; j < 16; ++j) {
            int idx = j * 256 + tid;           // 16-byte unit index, 4096 total
            int r = idx >> 6, c4 = (idx & 63) * 4;
            async_copy_b128(xg + r * DD + c4, &Xf[r][c4]);
        }
        async_wait0();
        __syncthreads();
#pragma unroll
        for (int j = 0; j < 16; ++j) {
            int idx = j * 256 + tid;
            int r = idx >> 8, c = idx & 255;
            Xh[r][c] = (half_t)Xf[r][c];
        }
    }
#else
#pragma unroll
    for (int j = 0; j < 16; ++j) {
        int idx = j * 256 + tid;
        int r = idx >> 8, c = idx & 255;
        float v = X[(t0 + r) * DD + c];
        Xf[r][c] = v;
        Xh[r][c] = (half_t)v;
    }
#endif
    __syncthreads();

    // ---- routing logits in fp32 (exact argmax) : 16 tokens x (64+16) outs ----
#pragma unroll
    for (int j = 0; j < 5; ++j) {
        int idx = tid * 5 + j;                   // 256*5 == 16*80 exactly
        int tok = idx / 80, o = idx % 80;
        const float* w = (o < GG) ? (Wgroup + o) : (Wslot + (o - GG));
        int stride = (o < GG) ? GG : KK;
        float s = 0.f;
        for (int k = 0; k < DD; ++k) s += Xf[tok][k] * w[(size_t)k * stride];
        Buf[tok][o] = s;
    }
    __syncthreads();
    if (tid < 16) {
        int gbest = 0; float gmax = Buf[tid][0];
        for (int o = 1; o < GG; ++o) { float v = Buf[tid][o]; if (v > gmax) { gmax = v; gbest = o; } }
        int sbest = 0; float smax = Buf[tid][GG];
        for (int o = 1; o < KK; ++o) { float v = Buf[tid][GG + o]; if (v > smax) { smax = v; sbest = o; } }
        slotv[tid] = gbest * KK + sbest;
    }
    __syncthreads();

    // ---- stage A: H1 = gelu(X @ Wm1 + bm1), 16x512, f16 out ----
#pragma unroll
    for (int t = 0; t < 4; ++t) {
        int n0 = (wave * 4 + t) * 16;
        __builtin_prefetch(Wm1t + (size_t)(n0 + arow) * DD, 0, 3);
        v8f acc = {};
        for (int kk = 0; kk < 8; ++kk) {
            v16h a  = load_a_frag(&Xh[0][0], DD + 16, kk * 32, lane);
            v16h bf = load_b_frag(Wm1t, DD, n0, kk * 32, lane);
            acc = __builtin_amdgcn_wmma_f32_16x16x32_f16(false, a, false, bf, (short)0, acc, false, false);
        }
        int col = n0 + arow;
#pragma unroll
        for (int e = 0; e < 8; ++e)
            H1h[e + (hi << 3)][col] = (half_t)gelu_exact(acc[e] + bm1[col]);
    }
    __syncthreads();

    // ---- stage B: msg = H1 @ Wm2 + bm2, 16x256 fp32 into Buf ----
#pragma unroll
    for (int t = 0; t < 2; ++t) {
        int n0 = (wave * 2 + t) * 16;
        __builtin_prefetch(Wm2t + (size_t)(n0 + arow) * D2, 0, 3);
        v8f acc = {};
        for (int kk = 0; kk < 16; ++kk) {
            v16h a  = load_a_frag(&H1h[0][0], D2 + 16, kk * 32, lane);
            v16h bf = load_b_frag(Wm2t, D2, n0, kk * 32, lane);
            acc = __builtin_amdgcn_wmma_f32_16x16x32_f16(false, a, false, bf, (short)0, acc, false, false);
        }
        int col = n0 + arow;
#pragma unroll
        for (int e = 0; e < 8; ++e)
            Buf[e + (hi << 3)][col] = acc[e] + bm2[col];
    }
    __syncthreads();

    // ---- stage C: Hg = gelu(X @ Wg1 + bg1), 16x256 fp32 reusing Xf ----
#pragma unroll
    for (int t = 0; t < 2; ++t) {
        int n0 = (wave * 2 + t) * 16;
        __builtin_prefetch(Wg1t + (size_t)(n0 + arow) * DD, 0, 3);
        v8f acc = {};
        for (int kk = 0; kk < 8; ++kk) {
            v16h a  = load_a_frag(&Xh[0][0], DD + 16, kk * 32, lane);
            v16h bf = load_b_frag(Wg1t, DD, n0, kk * 32, lane);
            acc = __builtin_amdgcn_wmma_f32_16x16x32_f16(false, a, false, bf, (short)0, acc, false, false);
        }
        int col = n0 + arow;
#pragma unroll
        for (int e = 0; e < 8; ++e)
            Xf[e + (hi << 3)][col] = gelu_exact(acc[e] + bg1[col]);
    }
    __syncthreads();

    // ---- gate = sigmoid(Hg @ Wg2 + bg2) ----
    {
        int tok = tid >> 4, part = tid & 15;
        float s = 0.f;
#pragma unroll
        for (int j = 0; j < 16; ++j) { int c = part * 16 + j; s += Xf[tok][c] * Wg2[c]; }
        red[tok][part] = s;
    }
    __syncthreads();
    if (tid < 16) {
        float s = 0.f;
        for (int p = 0; p < 16; ++p) s += red[tid][p];
        s += bg2[0];
        gatev[tid] = 1.f / (1.f + expf(-s));
    }
    __syncthreads();

    // ---- scatter: incoming[b, slot, :] += msg * gate (fp32 atomics) ----
    {
        int tok = tid >> 4, part = tid & 15;
        float g = gatev[tok];
        float* dst = incoming + ((size_t)b * NN + slotv[tok]) * DD;
#pragma unroll
        for (int j = 0; j < 16; ++j) {
            int c = part * 16 + j;
            __hip_atomic_fetch_add(dst + c, Buf[tok][c] * g,
                                   __ATOMIC_RELAXED, __HIP_MEMORY_SCOPE_AGENT);
        }
    }
}

__global__ __launch_bounds__(256)
void wta_slot_kernel(const float* __restrict__ S, const float* __restrict__ incoming,
                     const float* __restrict__ bu1, const float* __restrict__ bu2,
                     const float* __restrict__ ln_w, const float* __restrict__ ln_b,
                     const half_t* __restrict__ Wu1t, const half_t* __restrict__ Wu2t,
                     float* __restrict__ out) {
    __shared__ half_t Ah[16][D2 + 16];   // [S | incoming] f16
    __shared__ half_t Uh[16][D2 + 16];   // staging for incoming (fp32), then gelu(h@Wu1+bu1) f16
    __shared__ float  Sn[16][DD];        // S (residual), then S_new fp32
    __shared__ float  reds[16][17];
    __shared__ float  redq[16][17];

    const int tid  = threadIdx.x;
    const int wave = tid >> 5, lane = tid & 31;
    const int arow = lane & 15, hi = lane >> 4;
    const size_t r0 = (size_t)blockIdx.x * 16;   // 16 slot-rows; blocks tile B*N linearly

    // ---- stage S and incoming into LDS, build A = [S | incoming] in f16 ----
    float* If = (float*)&Uh[0][0];               // 16x256 fp32 fits in Uh region
#if HAS_ASYNC_LDS
    {
        const float* sg = S + r0 * DD;
        const float* ig = incoming + r0 * DD;
#pragma unroll
        for (int j = 0; j < 16; ++j) {
            int idx = j * 256 + tid;             // 16-byte unit index, 4096 per tensor
            int r = idx >> 6, c4 = (idx & 63) * 4;
            async_copy_b128(sg + r * DD + c4, &Sn[r][c4]);
            async_copy_b128(ig + r * DD + c4, If + r * DD + c4);
        }
        async_wait0();
        __syncthreads();
#pragma unroll
        for (int j = 0; j < 16; ++j) {
            int idx = j * 256 + tid;
            int r = idx >> 8, c = idx & 255;
            Ah[r][c]      = (half_t)Sn[r][c];
            Ah[r][DD + c] = (half_t)If[r * DD + c];
        }
    }
#else
#pragma unroll
    for (int j = 0; j < 16; ++j) {
        int idx = j * 256 + tid;
        int r = idx >> 8, c = idx & 255;
        float sv = S[(r0 + r) * DD + c];
        float iv = incoming[(r0 + r) * DD + c];
        Sn[r][c]      = sv;
        Ah[r][c]      = (half_t)sv;
        Ah[r][DD + c] = (half_t)iv;
    }
#endif
    __syncthreads();

    // ---- U = gelu(h @ Wu1 + bu1), 16x512 (overwrites Uh staging) ----
#pragma unroll
    for (int t = 0; t < 4; ++t) {
        int n0 = (wave * 4 + t) * 16;
        __builtin_prefetch(Wu1t + (size_t)(n0 + arow) * D2, 0, 3);
        v8f acc = {};
        for (int kk = 0; kk < 16; ++kk) {
            v16h a  = load_a_frag(&Ah[0][0], D2 + 16, kk * 32, lane);
            v16h bf = load_b_frag(Wu1t, D2, n0, kk * 32, lane);
            acc = __builtin_amdgcn_wmma_f32_16x16x32_f16(false, a, false, bf, (short)0, acc, false, false);
        }
        int col = n0 + arow;
#pragma unroll
        for (int e = 0; e < 8; ++e)
            Uh[e + (hi << 3)][col] = (half_t)gelu_exact(acc[e] + bu1[col]);
    }
    __syncthreads();

    // ---- S_new = S + U @ Wu2 + bu2 (S read from LDS Sn, overwritten in place) ----
#pragma unroll
    for (int t = 0; t < 2; ++t) {
        int n0 = (wave * 2 + t) * 16;
        __builtin_prefetch(Wu2t + (size_t)(n0 + arow) * D2, 0, 3);
        v8f acc = {};
        for (int kk = 0; kk < 16; ++kk) {
            v16h a  = load_a_frag(&Uh[0][0], D2 + 16, kk * 32, lane);
            v16h bf = load_b_frag(Wu2t, D2, n0, kk * 32, lane);
            acc = __builtin_amdgcn_wmma_f32_16x16x32_f16(false, a, false, bf, (short)0, acc, false, false);
        }
        int col = n0 + arow;
#pragma unroll
        for (int e = 0; e < 8; ++e) {
            int row = e + (hi << 3);
            Sn[row][col] = Sn[row][col] + acc[e] + bu2[col];   // same lane reads+writes its own element
        }
    }
    __syncthreads();

    // ---- LayerNorm over D=256 ----
    {
        int r = tid >> 4, part = tid & 15;
        float s = 0.f, q = 0.f;
#pragma unroll
        for (int j = 0; j < 16; ++j) { float v = Sn[r][part * 16 + j]; s += v; q += v * v; }
        reds[r][part] = s; redq[r][part] = q;
    }
    __syncthreads();
    {
        int r = tid >> 4, part = tid & 15;
        float s = 0.f, q = 0.f;
        for (int p = 0; p < 16; ++p) { s += reds[r][p]; q += redq[r][p]; }
        float mu  = s * (1.f / DD);
        float var = q * (1.f / DD) - mu * mu;
        float inv = rsqrtf(var + 1e-5f);
#pragma unroll
        for (int j = 0; j < 16; ++j) {
            int c = part * 16 + j;
            out[(r0 + r) * DD + c] = (Sn[r][c] - mu) * inv * ln_w[c] + ln_b[c];
        }
    }
}

extern "C" void kernel_launch(void* const* d_in, const int* in_sizes, int n_in,
                              void* d_out, int out_size, void* d_ws, size_t ws_size,
                              hipStream_t stream) {
    const float* X      = (const float*)d_in[0];
    const float* S      = (const float*)d_in[1];
    const float* Wgroup = (const float*)d_in[2];
    const float* Wslot  = (const float*)d_in[3];
    const float* Wm1    = (const float*)d_in[4];
    const float* bm1    = (const float*)d_in[5];
    const float* Wm2    = (const float*)d_in[6];
    const float* bm2    = (const float*)d_in[7];
    const float* Wg1    = (const float*)d_in[8];
    const float* bg1    = (const float*)d_in[9];
    const float* Wg2    = (const float*)d_in[10];
    const float* bg2    = (const float*)d_in[11];
    const float* Wu1    = (const float*)d_in[12];
    const float* bu1    = (const float*)d_in[13];
    const float* Wu2    = (const float*)d_in[14];
    const float* bu2    = (const float*)d_in[15];
    const float* ln_w   = (const float*)d_in[16];
    const float* ln_b   = (const float*)d_in[17];

    char* ws = (char*)d_ws;
    float* incoming = (float*)ws;                                    // B*N*D fp32 = 8 MB
    size_t off = (size_t)BB * NN * DD * sizeof(float);
    half_t* Wm1t = (half_t*)(ws + off); off += (size_t)DD * D2 * sizeof(half_t);
    half_t* Wm2t = (half_t*)(ws + off); off += (size_t)D2 * DD * sizeof(half_t);
    half_t* Wg1t = (half_t*)(ws + off); off += (size_t)DD * DD * sizeof(half_t);
    half_t* Wu1t = (half_t*)(ws + off); off += (size_t)D2 * D2 * sizeof(half_t);
    half_t* Wu2t = (half_t*)(ws + off); off += (size_t)D2 * DD * sizeof(half_t);

    // zero the scatter accumulator every call (ws is poisoned / stale)
    (void)hipMemsetAsync(incoming, 0, (size_t)BB * NN * DD * sizeof(float), stream);

    int n;
    n = DD * D2; convw_kernel<<<(n + 255) / 256, 256, 0, stream>>>(Wm1, Wm1t, DD, D2);
    n = D2 * DD; convw_kernel<<<(n + 255) / 256, 256, 0, stream>>>(Wm2, Wm2t, D2, DD);
    n = DD * DD; convw_kernel<<<(n + 255) / 256, 256, 0, stream>>>(Wg1, Wg1t, DD, DD);
    n = D2 * D2; convw_kernel<<<(n + 255) / 256, 256, 0, stream>>>(Wu1, Wu1t, D2, D2);
    n = D2 * DD; convw_kernel<<<(n + 255) / 256, 256, 0, stream>>>(Wu2, Wu2t, D2, DD);

    wta_token_kernel<<<(BB * LL) / 16, 256, 0, stream>>>(
        X, Wgroup, Wslot, bm1, bm2, bg1, bg2, Wg2, Wm1t, Wm2t, Wg1t, incoming);

    wta_slot_kernel<<<(BB * NN) / 16, 256, 0, stream>>>(
        S, incoming, bu1, bu2, ln_w, ln_b, Wu1t, Wu2t, (float*)d_out);
}
